// AADecoderBlock_4063039062778
// MI455X (gfx1250) — matrix operand order repaired and checked by reading, not compiled, not run on hardware
//
#include <hip/hip_runtime.h>
#include <hip/hip_bf16.h>

// ---------------------------------------------------------------------------
// Types for CDNA5 WMMA (wave32, 16x16x32 bf16 -> f32)
// ---------------------------------------------------------------------------
typedef __bf16 bf16;
typedef __attribute__((ext_vector_type(16))) __bf16 v16bf;
typedef __attribute__((ext_vector_type(8)))  float  v8f;
typedef __attribute__((ext_vector_type(8)))  unsigned int v8u;
typedef __attribute__((ext_vector_type(4)))  int vi4;

union FragBF { v16bf h; v8u u; };

#define N_RES 16384
#define KNBR  32

// CDNA5 async global->LDS copy path (ASYNCcnt), guarded so either toolchain builds.
#if defined(__AMDGCN__) && __has_builtin(__builtin_amdgcn_global_load_async_to_lds_b128) && __has_builtin(__builtin_amdgcn_s_wait_asynccnt)
#define ASYNC_LDS 1
#endif

__device__ inline void cp_b128(const bf16* g, bf16* l) {
#ifdef ASYNC_LDS
  __builtin_amdgcn_global_load_async_to_lds_b128(
      (__attribute__((address_space(1))) vi4*)(bf16*)g,
      (__attribute__((address_space(3))) vi4*)l, 0, 0);
#else
  *(float4*)l = *(const float4*)g;
#endif
}

template <int PENDING>
__device__ inline void wait_cp() {
#ifdef ASYNC_LDS
  __builtin_amdgcn_s_wait_asynccnt(PENDING);
#endif
}

__device__ inline v8f wmma_bf16(v16bf a, v16bf b, v8f c) {
  return __builtin_amdgcn_wmma_f32_16x16x32_bf16(false, a, false, b, (short)0, c,
                                                 false, false);
}

// A fragment: 16 rows x 32 K, row-major source with leading dim ld.
// ISA layout: lane m=l&15 row M=m; half=l>>4; VGPR j<4: K = half*8+2j,
// VGPR j>=4: K = 16+half*8+2(j-4). Pairs contiguous -> two b128 loads.
__device__ inline v16bf load_a32(const bf16* p, int ld) {
  int lane = threadIdx.x & 31;
  int m = lane & 15, half = lane >> 4;
  const bf16* r = p + m * ld + half * 8;
  FragBF f;
#pragma unroll
  for (int j = 0; j < 8; ++j) {
    int k0 = (j < 4) ? (2 * j) : (16 + 2 * (j - 4));
    f.u[j] = *(const unsigned int*)(r + k0);
  }
  return f.h;
}

// B fragment: 32 K x 16 N, from column-major (N-major) store Bt[N][K], ld = K.
__device__ inline v16bf load_b32(const bf16* p, int ld) {
  int lane = threadIdx.x & 31;
  int nn = lane & 15, half = lane >> 4;
  const bf16* r = p + nn * ld + half * 8;
  FragBF f;
#pragma unroll
  for (int j = 0; j < 8; ++j) f.u[j] = *(const unsigned int*)(r + ((j < 4) ? (2 * j) : (16 + 2 * (j - 4))));
  return f.h;
}

__device__ inline float gelu_f(float x) {
  float x3 = x * x * x;
  return 0.5f * x * (1.f + tanhf(0.7978845608028654f * (x + 0.044715f * x3)));
}

// ---------------------------------------------------------------------------
// Weight transpose fp32 -> bf16 column-major  (Bt[c*K + k] = W[k*Ncol + c])
// ---------------------------------------------------------------------------
__global__ void transpose_bf16_kernel(const float* __restrict__ W,
                                      bf16* __restrict__ out, int K, int Ncol) {
  int idx = blockIdx.x * blockDim.x + threadIdx.x;
  if (idx >= K * Ncol) return;
  int k = idx / Ncol, c = idx - k * Ncol;
  out[c * K + k] = (bf16)W[idx];
}

// W_comb = W_mlp2(128x64) @ Wb(64x8), padded to 16 cols, stored col-major
// [16][128]; b_comb = b_mlp2 @ Wb (8).
__global__ void wcomb_kernel(const float* __restrict__ Wm2,
                             const float* __restrict__ Wb,
                             const float* __restrict__ b2,
                             bf16* __restrict__ Wcomb_t,
                             float* __restrict__ b_comb) {
  int idx = blockIdx.x * blockDim.x + threadIdx.x;
  if (idx >= 16 * 128) return;
  int nn = idx >> 7, kk = idx & 127;
  float acc = 0.f;
  if (nn < 8) {
    for (int p = 0; p < 64; ++p) acc += Wm2[kk * 64 + p] * Wb[p * 8 + nn];
  }
  Wcomb_t[nn * 128 + kk] = (bf16)acc;
  if (idx < 8) {
    float bb = 0.f;
    for (int p = 0; p < 64; ++p) bb += b2[p] * Wb[p * 8 + idx];
    b_comb[idx] = bb;
  }
}

// ---------------------------------------------------------------------------
// Frames: R (rows x [e1 e2 e3] columns), t=CA, local_pos (15), aa embedding LN
// ---------------------------------------------------------------------------
__global__ void frames_kernel(const float* __restrict__ pos,
                              const int* __restrict__ aa,
                              const float* __restrict__ W_aa,
                              const float* __restrict__ aas,
                              const float* __restrict__ aab,
                              float* __restrict__ R, float* __restrict__ t,
                              float* __restrict__ lp,
                              float* __restrict__ aaE) {
  int n = blockIdx.x * blockDim.x + threadIdx.x;
  if (n >= N_RES) return;
  const float* pp = pos + n * 15;
  float Np[3], ca[3], cc[3];
#pragma unroll
  for (int i = 0; i < 3; ++i) { Np[i] = pp[i]; ca[i] = pp[3 + i]; cc[i] = pp[6 + i]; }
  float e1[3], e2[3], e3[3], u[3];
#pragma unroll
  for (int i = 0; i < 3; ++i) e1[i] = cc[i] - ca[i];
  float n1 = sqrtf(e1[0]*e1[0] + e1[1]*e1[1] + e1[2]*e1[2]);
  float iv = 1.f / (n1 + 1e-8f);
#pragma unroll
  for (int i = 0; i < 3; ++i) e1[i] *= iv;
#pragma unroll
  for (int i = 0; i < 3; ++i) u[i] = Np[i] - ca[i];
  float du = u[0]*e1[0] + u[1]*e1[1] + u[2]*e1[2];
#pragma unroll
  for (int i = 0; i < 3; ++i) u[i] -= du * e1[i];
  float n2 = sqrtf(u[0]*u[0] + u[1]*u[1] + u[2]*u[2]);
  float iv2 = 1.f / (n2 + 1e-8f);
#pragma unroll
  for (int i = 0; i < 3; ++i) e2[i] = u[i] * iv2;
  e3[0] = e1[1]*e2[2] - e1[2]*e2[1];
  e3[1] = e1[2]*e2[0] - e1[0]*e2[2];
  e3[2] = e1[0]*e2[1] - e1[1]*e2[0];
  float Rl[9];
#pragma unroll
  for (int i = 0; i < 3; ++i) { Rl[i*3+0] = e1[i]; Rl[i*3+1] = e2[i]; Rl[i*3+2] = e3[i]; }
#pragma unroll
  for (int i = 0; i < 9; ++i) R[n*9 + i] = Rl[i];
#pragma unroll
  for (int i = 0; i < 3; ++i) t[n*3 + i] = ca[i];
#pragma unroll
  for (int a = 0; a < 5; ++a)
#pragma unroll
    for (int i = 0; i < 3; ++i) {
      float s = 0.f;
#pragma unroll
      for (int j = 0; j < 3; ++j) s += Rl[j*3+i] * (pp[a*3+j] - ca[j]);
      lp[n*15 + a*3 + i] = s;
    }
  const float* w = W_aa + aa[n] * 64;
  float m = 0.f;
  for (int c = 0; c < 64; ++c) m += w[c];
  m *= (1.f / 64.f);
  float v = 0.f;
  for (int c = 0; c < 64; ++c) { float d = w[c] - m; v += d * d; }
  v *= (1.f / 64.f);
  float rs = rsqrtf(v + 1e-5f);
  for (int c = 0; c < 64; ++c) aaE[n*64 + c] = (w[c] - m) * rs * aas[c] + aab[c];
}

// ---------------------------------------------------------------------------
// LayerNorm, D=512, one wave per row
// ---------------------------------------------------------------------------
__global__ __launch_bounds__(256) void ln512_kernel(
    const float* __restrict__ X, const float* __restrict__ s,
    const float* __restrict__ b, float* __restrict__ outF,
    bf16* __restrict__ outB, int rows) {
  int row = blockIdx.x * 8 + (threadIdx.x >> 5);
  int lane = threadIdx.x & 31;
  if (row >= rows) return;
  const float* x = X + (size_t)row * 512;
  float v[16];
  float sum = 0.f;
#pragma unroll
  for (int i = 0; i < 16; ++i) { v[i] = x[lane + i*32]; sum += v[i]; }
#pragma unroll
  for (int o = 16; o; o >>= 1) sum += __shfl_xor(sum, o, 32);
  float mean = sum * (1.f / 512.f);
  float var = 0.f;
#pragma unroll
  for (int i = 0; i < 16; ++i) { float d = v[i] - mean; var += d * d; }
#pragma unroll
  for (int o = 16; o; o >>= 1) var += __shfl_xor(var, o, 32);
  var *= (1.f / 512.f);
  float rs = rsqrtf(var + 1e-5f);
#pragma unroll
  for (int i = 0; i < 16; ++i) {
    int idx = lane + i * 32;
    float y = (v[i] - mean) * rs * s[idx] + b[idx];
    if (outF) outF[(size_t)row*512 + idx] = y;
    if (outB) outB[(size_t)row*512 + idx] = (bf16)y;
  }
}

// ---------------------------------------------------------------------------
// Generic bf16 WMMA GEMM with double-buffered async B staging through LDS.
// C[MxN] = A[MxK] * Bt[NxK]^T (+bias) (+resid)
// block = 256 thr = 8 waves; wave owns 16 rows x 64 cols (4 wmma tiles).
// B strip (64 cols x 32 K = 4KB) staged per k-step: issue s+1, wait s, barrier.
// ---------------------------------------------------------------------------
template <bool BIAS, bool RES, bool OUTF, bool OUTB>
__global__ __launch_bounds__(256) void gemm_kernel(
    const bf16* __restrict__ A, const bf16* __restrict__ Bt,
    const float* __restrict__ bias, const float* __restrict__ resid,
    float* __restrict__ Cf, bf16* __restrict__ Cb, int M, int Nc, int Kc) {
  __shared__ __align__(16) bf16 Bs[2][64][32];
  int tid = threadIdx.x;
  int wave = tid >> 5;
  int lane = tid & 31;
  int rowBase = blockIdx.y * 128 + wave * 16;
  int colBase = blockIdx.x * 64;
  // staging decomposition: 256 thr x 16B == 4KB strip
  int scol = tid >> 2, skp = (tid & 3) * 8;
  const bf16* gB = Bt + (size_t)(colBase + scol) * Kc + skp;
  v8f acc[4] = {v8f{}, v8f{}, v8f{}, v8f{}};
  int nsteps = Kc >> 5;
  cp_b128(gB, &Bs[0][scol][skp]);
  for (int s = 0; s < nsteps; ++s) {
    int cur = s & 1;
    if (s + 1 < nsteps) {
      cp_b128(gB + (s + 1) * 32, &Bs[cur ^ 1][scol][skp]);
      wait_cp<1>();
    } else {
      wait_cp<0>();
    }
    __syncthreads();
    v16bf a = load_a32(A + (size_t)rowBase * Kc + s * 32, Kc);
#pragma unroll
    for (int c = 0; c < 4; ++c) {
      v16bf b = load_b32(&Bs[cur][c * 16][0], 32);
      acc[c] = wmma_bf16(a, b, acc[c]);
    }
    __syncthreads();
  }
  int nIdx = lane & 15, half = lane >> 4;
#pragma unroll
  for (int c = 0; c < 4; ++c) {
    int col = colBase + 16*c + nIdx;
    float bv = BIAS ? bias[col] : 0.f;
#pragma unroll
    for (int r = 0; r < 8; ++r) {
      int row = rowBase + half * 8 + r;
      float v = acc[c][r] + bv;
      if (RES) v += resid[(size_t)row * Nc + col];
      if (OUTF) Cf[(size_t)row * Nc + col] = v;
      if (OUTB) Cb[(size_t)row * Nc + col] = (bf16)v;
    }
  }
}

// Dual GEMM: u = y@Wu, g = y@Wg, out = gelu(g)*u (bf16); async-staged B pair.
__global__ __launch_bounds__(256) void gemm_gu_kernel(
    const bf16* __restrict__ A, const bf16* __restrict__ But,
    const bf16* __restrict__ Bgt, bf16* __restrict__ gu, int M, int Nc, int Kc) {
  __shared__ __align__(16) bf16 Bs[2][2][64][32];
  int tid = threadIdx.x;
  int wave = tid >> 5;
  int lane = tid & 31;
  int rowBase = blockIdx.y * 128 + wave * 16;
  int colBase = blockIdx.x * 64;
  int scol = tid >> 2, skp = (tid & 3) * 8;
  const bf16* gU = But + (size_t)(colBase + scol) * Kc + skp;
  const bf16* gG = Bgt + (size_t)(colBase + scol) * Kc + skp;
  v8f au[4] = {v8f{}, v8f{}, v8f{}, v8f{}};
  v8f ag[4] = {v8f{}, v8f{}, v8f{}, v8f{}};
  int nsteps = Kc >> 5;
  cp_b128(gU, &Bs[0][0][scol][skp]);
  cp_b128(gG, &Bs[0][1][scol][skp]);
  for (int s = 0; s < nsteps; ++s) {
    int cur = s & 1;
    if (s + 1 < nsteps) {
      cp_b128(gU + (s + 1) * 32, &Bs[cur ^ 1][0][scol][skp]);
      cp_b128(gG + (s + 1) * 32, &Bs[cur ^ 1][1][scol][skp]);
      wait_cp<2>();
    } else {
      wait_cp<0>();
    }
    __syncthreads();
    v16bf a = load_a32(A + (size_t)rowBase * Kc + s * 32, Kc);
#pragma unroll
    for (int c = 0; c < 4; ++c) {
      v16bf bu = load_b32(&Bs[cur][0][c * 16][0], 32);
      v16bf bg = load_b32(&Bs[cur][1][c * 16][0], 32);
      au[c] = wmma_bf16(a, bu, au[c]);
      ag[c] = wmma_bf16(a, bg, ag[c]);
    }
    __syncthreads();
  }
  int nIdx = lane & 15, half = lane >> 4;
#pragma unroll
  for (int c = 0; c < 4; ++c) {
    int col = colBase + 16*c + nIdx;
#pragma unroll
    for (int r = 0; r < 8; ++r) {
      int row = rowBase + half * 8 + r;
      gu[(size_t)row * Nc + col] = (bf16)(gelu_f(ag[c][r]) * au[c][r]);
    }
  }
}

// ---------------------------------------------------------------------------
// Fused pair features + LN + MLP1 (WMMA) + (MLP2@Wb folded, WMMA) -> bias
// one block per residue; 32 pairs x 64 channels. Weights async-preloaded to
// LDS at block start, overlapped with the geometry phase.
// ---------------------------------------------------------------------------
__global__ __launch_bounds__(256) void pair_bias_kernel(
    const int* __restrict__ nbr, const int* __restrict__ resi,
    const int* __restrict__ chain, const int* __restrict__ batch,
    const float* __restrict__ R, const float* __restrict__ t,
    const float* __restrict__ pos, const float* __restrict__ aaE,
    const float* __restrict__ W_relpos, const float* __restrict__ W_dist,
    const float* __restrict__ W_dir, const float* __restrict__ W_rot,
    const float* __restrict__ W_pairvec, const float* __restrict__ ln_s,
    const float* __restrict__ ln_b, const bf16* __restrict__ Wmlp1_t,
    const float* __restrict__ b_mlp1, const bf16* __restrict__ Wcomb_t,
    const float* __restrict__ b_comb, float* __restrict__ bias_out) {
  int n = blockIdx.x;
  int tid = threadIdx.x;
  __shared__ float geo[32][48];      // rbf[0..15], dlocn[16..18], rotf[19..30], loc[31..45]
  __shared__ int   gnb[32];
  __shared__ int   grel[32];
  __shared__ float gsame[32];
  __shared__ float pairAcc[32][64];
  __shared__ __align__(16) bf16 aBuf[32][64];
  __shared__ __align__(16) bf16 hBuf[32][128];
  __shared__ __align__(16) bf16 Wm1s[128][64];   // 16KB, col-major [col][k]
  __shared__ __align__(16) bf16 Wcbs[16][128];   //  4KB, col-major [col][k]

  // async-prefetch MLP weights into LDS (5 issues/thread), overlap with geometry
#pragma unroll
  for (int i = 0; i < 4; ++i) {
    int chunk = tid + i * 256;           // 1024 x 16B = 16KB
    cp_b128(Wmlp1_t + chunk * 8, &Wm1s[0][0] + chunk * 8);
  }
  cp_b128(Wcomb_t + tid * 8, &Wcbs[0][0] + tid * 8);  // 256 x 16B = 4KB

  if (tid < 32) {
    int k = tid;
    int j = nbr[n * 32 + k];
    int nb = (j != -1) ? j : 0;
    gnb[k] = nb;
    int dr = resi[nb] - resi[n];
    dr = min(max(dr, -32), 32) + 32;
    grel[k] = dr;
    gsame[k] = ((chain[nb] == chain[n]) && (batch[nb] == batch[n])) ? 1.f : 0.f;
    float Rn[9], Rj[9], tn[3], tj[3];
#pragma unroll
    for (int i = 0; i < 9; ++i) { Rn[i] = R[n*9+i]; Rj[i] = R[(size_t)nb*9+i]; }
#pragma unroll
    for (int i = 0; i < 3; ++i) { tn[i] = t[n*3+i]; tj[i] = t[(size_t)nb*3+i]; }
    float dv[3];
#pragma unroll
    for (int i = 0; i < 3; ++i) dv[i] = tj[i] - tn[i];
    float d = sqrtf(dv[0]*dv[0] + dv[1]*dv[1] + dv[2]*dv[2]);
#pragma unroll
    for (int b = 0; b < 16; ++b) {
      float bin = 22.f * (float)b / 15.f;
      float z = (d - bin) * (1.f / 1.375f);
      geo[k][b] = __expf(-z * z);
    }
    float dl[3];
#pragma unroll
    for (int i = 0; i < 3; ++i) {
      float s = 0.f;
#pragma unroll
      for (int jj = 0; jj < 3; ++jj) s += Rn[jj*3+i] * dv[jj];
      dl[i] = s;
    }
    float dn = sqrtf(dl[0]*dl[0] + dl[1]*dl[1] + dl[2]*dl[2]);
    float idn = 1.f / (dn + 1e-8f);
#pragma unroll
    for (int i = 0; i < 3; ++i) geo[k][16 + i] = dl[i] * idn;
#pragma unroll
    for (int i = 0; i < 3; ++i)
#pragma unroll
      for (int l = 0; l < 3; ++l) {
        float s = 0.f;
#pragma unroll
        for (int jj = 0; jj < 3; ++jj) s += Rn[jj*3+i] * Rj[jj*3+l];
        geo[k][19 + i*3 + l] = s;
      }
#pragma unroll
    for (int i = 0; i < 3; ++i) geo[k][28 + i] = dl[i];   // relT at rotf[9..11]
    const float* pj = pos + (size_t)nb * 15;
#pragma unroll
    for (int a = 0; a < 5; ++a)
#pragma unroll
      for (int i = 0; i < 3; ++i) {
        float s = 0.f;
#pragma unroll
        for (int jj = 0; jj < 3; ++jj) s += Rn[jj*3+i] * (pj[a*3+jj] - tn[jj]);
        geo[k][31 + a*3 + i] = s;
      }
  }
  __syncthreads();

  // projections: each thread does one pair x 8 channels
  {
    int p = tid >> 3, c0 = (tid & 7) * 8;
    int rel = grel[p];
    float same = gsame[p];
    float a8[8];
#pragma unroll
    for (int c = 0; c < 8; ++c) a8[c] = same * W_relpos[rel*64 + c0 + c];
    for (int b = 0; b < 16; ++b) {
      float r = geo[p][b];
#pragma unroll
      for (int c = 0; c < 8; ++c) a8[c] += r * W_dist[b*64 + c0 + c];
    }
    for (int i = 0; i < 3; ++i) {
      float r = geo[p][16 + i];
#pragma unroll
      for (int c = 0; c < 8; ++c) a8[c] += r * W_dir[i*64 + c0 + c];
    }
    for (int i = 0; i < 12; ++i) {
      float r = geo[p][19 + i];
#pragma unroll
      for (int c = 0; c < 8; ++c) a8[c] += r * W_rot[i*64 + c0 + c];
    }
    for (int i = 0; i < 15; ++i) {
      float r = geo[p][31 + i];
#pragma unroll
      for (int c = 0; c < 8; ++c) a8[c] += r * W_pairvec[i*64 + c0 + c];
    }
#pragma unroll
    for (int c = 0; c < 8; ++c) pairAcc[p][c0 + c] = a8[c];
  }
  __syncthreads();

  // LN over 64 channels + aa_emb[nb], produce bf16 A matrix
  if (tid < 32) {
    int p = tid;
    float m = 0.f;
    for (int c = 0; c < 64; ++c) m += pairAcc[p][c];
    m *= (1.f / 64.f);
    float v = 0.f;
    for (int c = 0; c < 64; ++c) { float d = pairAcc[p][c] - m; v += d * d; }
    v *= (1.f / 64.f);
    float rs = rsqrtf(v + 1e-5f);
    const float* ae = aaE + (size_t)gnb[p] * 64;
    for (int c = 0; c < 64; ++c) {
      float y = (pairAcc[p][c] - m) * rs * ln_s[c] + ln_b[c] + ae[c];
      aBuf[p][c] = (bf16)y;
    }
  }
  wait_cp<0>();       // weights resident in LDS before WMMA stages
  __syncthreads();

  // MLP1: [32x64] @ [64x128], gelu -> hBuf  (16 tiles over 8 waves)
  {
    int wave = tid >> 5, lane = tid & 31;
    int rt = wave & 1;
    int ct0 = wave >> 1;
    v8f acc0 = {}, acc1 = {};
    for (int kb = 0; kb < 64; kb += 32) {
      v16bf a = load_a32(&aBuf[rt*16][0] + kb, 64);
      v16bf b0 = load_b32(&Wm1s[ct0*16][0] + kb, 64);
      v16bf b1 = load_b32(&Wm1s[(ct0+4)*16][0] + kb, 64);
      acc0 = wmma_bf16(a, b0, acc0);
      acc1 = wmma_bf16(a, b1, acc1);
    }
    int nn = lane & 15, half = lane >> 4;
#pragma unroll
    for (int tix = 0; tix < 2; ++tix) {
      int ct = (tix == 0) ? ct0 : (ct0 + 4);
      v8f* ac = (tix == 0) ? &acc0 : &acc1;
      int col = ct * 16 + nn;
      float bv = b_mlp1[col];
#pragma unroll
      for (int r = 0; r < 8; ++r) {
        int row = rt * 16 + half * 8 + r;
        hBuf[row][col] = (bf16)gelu_f((*ac)[r] + bv);
      }
    }
  }
  __syncthreads();

  // hidden @ Wcomb (128x16, cols>=8 zero) + b_comb -> bias logits [32][8]
  {
    int wave = tid >> 5, lane = tid & 31;
    if (wave < 2) {
      int rt = wave;
      v8f acc = {};
      for (int kb = 0; kb < 128; kb += 32) {
        v16bf a = load_a32(&hBuf[rt*16][0] + kb, 128);
        v16bf b = load_b32(&Wcbs[0][0] + kb, 128);
        acc = wmma_bf16(a, b, acc);
      }
      int nn = lane & 15, half = lane >> 4;
      if (nn < 8) {
#pragma unroll
        for (int r = 0; r < 8; ++r) {
          int row = rt * 16 + half * 8 + r;
          bias_out[(size_t)n * 256 + row * 8 + nn] = acc[r] + b_comb[nn];
        }
      }
    }
  }
}

// ---------------------------------------------------------------------------
// Attention: one block per residue, one wave per head
// ---------------------------------------------------------------------------
__global__ __launch_bounds__(256) void attn_kernel(
    const int* __restrict__ nbr, const unsigned char* __restrict__ mask,
    const float* __restrict__ q, const float* __restrict__ kbuf,
    const float* __restrict__ vbuf, const float* __restrict__ bias,
    bf16* __restrict__ o_out) {
  int n = blockIdx.x, tid = threadIdx.x;
  __shared__ int   snb[32];
  __shared__ float spm[32];
  __shared__ float qs[256];
  __shared__ float satt[8][32];
  if (tid < 32) {
    int j = nbr[n * 32 + tid];
    bool valid = (j != -1);
    int nb = valid ? j : 0;
    snb[tid] = nb;
    spm[tid] = (valid && mask[n] && mask[nb]) ? 1.f : 0.f;
  }
  qs[tid] = q[(size_t)n * 256 + tid];
  __syncthreads();
  int h = tid >> 5, k = tid & 31;
  int kn = snb[k];
  const float* kp = kbuf + (size_t)kn * 256 + h * 32;
  float lg = 0.f;
#pragma unroll
  for (int d = 0; d < 32; ++d) lg += qs[h * 32 + d] * kp[d];
  lg = lg * 0.17677669529663687f + bias[(size_t)n * 256 + k * 8 + h];
  float pm = spm[k];
  if (pm <= 0.f) lg = -1e9f;
  float m = lg;
#pragma unroll
  for (int o = 16; o; o >>= 1) m = fmaxf(m, __shfl_xor(m, o, 32));
  float e = __expf(lg - m);
  float s = e;
#pragma unroll
  for (int o = 16; o; o >>= 1) s += __shfl_xor(s, o, 32);
  satt[h][k] = (e / s) * pm;
  __syncthreads();
  int d = k;
  float o = 0.f;
#pragma unroll
  for (int kk = 0; kk < 32; ++kk)
    o += satt[h][kk] * vbuf[(size_t)snb[kk] * 256 + h * 32 + d];
  o_out[(size_t)n * 256 + h * 32 + d] = (bf16)o;
}

// local_pos (15) @ W_lp1 (15x1024) + b, gelu -> bf16
__global__ void lp_mlp1_kernel(const float* __restrict__ lp,
                               const float* __restrict__ W1,
                               const float* __restrict__ b1,
                               bf16* __restrict__ out) {
  int idx = blockIdx.x * blockDim.x + threadIdx.x;
  if (idx >= N_RES * 1024) return;
  int n = idx >> 10, j = idx & 1023;
  const float* l = lp + n * 15;
  float acc = b1[j];
#pragma unroll
  for (int q = 0; q < 15; ++q) acc += l[q] * W1[q * 1024 + j];
  out[idx] = (bf16)gelu_f(acc);
}

// ---------------------------------------------------------------------------
extern "C" void kernel_launch(void* const* d_in, const int* in_sizes, int n_in,
                              void* d_out, int out_size, void* d_ws, size_t ws_size,
                              hipStream_t stream) {
  (void)in_sizes; (void)n_in; (void)out_size; (void)ws_size;
  const int*   aa       = (const int*)d_in[0];
  const float* features = (const float*)d_in[1];
  const float* pos      = (const float*)d_in[2];
  const int*   nbr      = (const int*)d_in[3];
  const int*   resi     = (const int*)d_in[4];
  const int*   chain    = (const int*)d_in[5];
  const int*   batch    = (const int*)d_in[6];
  const unsigned char* mask = (const unsigned char*)d_in[7];
  const float* W_relpos = (const float*)d_in[8];
  const float* W_dist   = (const float*)d_in[9];
  const float* W_dir    = (const float*)d_in[10];
  const float* W_rot    = (const float*)d_in[11];
  const float* W_pairvec= (const float*)d_in[12];
  const float* ln_pair_s= (const float*)d_in[13];
  const float* ln_pair_b= (const float*)d_in[14];
  const float* W_aa     = (const float*)d_in[15];
  const float* ln_aa_s  = (const float*)d_in[16];
  const float* ln_aa_b  = (const float*)d_in[17];
  const float* W_mlp1   = (const float*)d_in[18];
  const float* b_mlp1   = (const float*)d_in[19];
  const float* W_mlp2   = (const float*)d_in[20];
  const float* b_mlp2   = (const float*)d_in[21];
  const float* ln_att_s = (const float*)d_in[22];
  const float* ln_att_b = (const float*)d_in[23];
  const float* Wq       = (const float*)d_in[24];
  const float* Wk       = (const float*)d_in[25];
  const float* Wv       = (const float*)d_in[26];
  const float* Wb       = (const float*)d_in[27];
  const float* Wo       = (const float*)d_in[28];
  const float* ln_up_s  = (const float*)d_in[29];
  const float* ln_up_b  = (const float*)d_in[30];
  const float* W_lp1    = (const float*)d_in[31];
  const float* b_lp1    = (const float*)d_in[32];
  const float* W_lp2    = (const float*)d_in[33];
  const float* b_lp2    = (const float*)d_in[34];
  const float* Wu       = (const float*)d_in[35];
  const float* Wg       = (const float*)d_in[36];
  const float* Wo2      = (const float*)d_in[37];
  const float* b_o2     = (const float*)d_in[38];
  float* out = (float*)d_out;

  const size_t N = N_RES;
  size_t off = 0;
  auto alloc = [&](size_t bytes) -> void* {
    off = (off + 255) & ~(size_t)255;
    void* p = (void*)((char*)d_ws + off);
    off += bytes;
    return p;
  };
  float* R     = (float*)alloc(N * 9 * 4);
  float* t     = (float*)alloc(N * 3 * 4);
  float* lp    = (float*)alloc(N * 15 * 4);
  float* aaE   = (float*)alloc(N * 64 * 4);
  bf16*  xb    = (bf16*) alloc(N * 512 * 2);
  float* qB    = (float*)alloc(N * 256 * 4);
  float* kB    = (float*)alloc(N * 256 * 4);
  float* vB    = (float*)alloc(N * 256 * 4);
  bf16*  oB    = (bf16*) alloc(N * 256 * 2);
  float* biasL = (float*)alloc(N * 256 * 4);
  float* yln   = (float*)alloc(N * 512 * 4);
  bf16*  lph   = (bf16*) alloc(N * 1024 * 2);
  bf16*  yB    = (bf16*) alloc(N * 512 * 2);
  bf16*  guB   = (bf16*) alloc(N * 2048 * 2);
  bf16*  Wq_t  = (bf16*) alloc(512 * 256 * 2);
  bf16*  Wk_t  = (bf16*) alloc(512 * 256 * 2);
  bf16*  Wv_t  = (bf16*) alloc(512 * 256 * 2);
  bf16*  Wo_t  = (bf16*) alloc(256 * 512 * 2);
  bf16*  Wm1_t = (bf16*) alloc(64 * 128 * 2);
  bf16*  Wl2_t = (bf16*) alloc(1024 * 512 * 2);
  bf16*  Wu_t  = (bf16*) alloc(512 * 2048 * 2);
  bf16*  Wg_t  = (bf16*) alloc(512 * 2048 * 2);
  bf16*  Wo2_t = (bf16*) alloc(2048 * 512 * 2);
  bf16*  Wcb_t = (bf16*) alloc(16 * 128 * 2);
  float* b_cb  = (float*)alloc(8 * 4);

  auto tpose = [&](const float* W, bf16* o, int Kc, int Nc) {
    int tot = Kc * Nc;
    transpose_bf16_kernel<<<(tot + 255) / 256, 256, 0, stream>>>(W, o, Kc, Nc);
  };
  tpose(Wq, Wq_t, 512, 256);
  tpose(Wk, Wk_t, 512, 256);
  tpose(Wv, Wv_t, 512, 256);
  tpose(Wo, Wo_t, 256, 512);
  tpose(W_mlp1, Wm1_t, 64, 128);
  tpose(W_lp2, Wl2_t, 1024, 512);
  tpose(Wu, Wu_t, 512, 2048);
  tpose(Wg, Wg_t, 512, 2048);
  tpose(Wo2, Wo2_t, 2048, 512);
  wcomb_kernel<<<8, 256, 0, stream>>>(W_mlp2, Wb, b_mlp2, Wcb_t, b_cb);

  frames_kernel<<<(int)(N / 256), 256, 0, stream>>>(pos, aa, W_aa, ln_aa_s, ln_aa_b,
                                                    R, t, lp, aaE);
  ln512_kernel<<<(int)(N / 8), 256, 0, stream>>>(features, ln_att_s, ln_att_b,
                                                 nullptr, xb, (int)N);
  dim3 g256(256 / 64, (int)(N / 128));
  gemm_kernel<false, false, true, false><<<g256, 256, 0, stream>>>(
      xb, Wq_t, nullptr, nullptr, qB, nullptr, (int)N, 256, 512);
  gemm_kernel<false, false, true, false><<<g256, 256, 0, stream>>>(
      xb, Wk_t, nullptr, nullptr, kB, nullptr, (int)N, 256, 512);
  gemm_kernel<false, false, true, false><<<g256, 256, 0, stream>>>(
      xb, Wv_t, nullptr, nullptr, vB, nullptr, (int)N, 256, 512);
  pair_bias_kernel<<<(int)N, 256, 0, stream>>>(
      nbr, resi, chain, batch, R, t, pos, aaE, W_relpos, W_dist, W_dir, W_rot,
      W_pairvec, ln_pair_s, ln_pair_b, Wm1_t, b_mlp1, Wcb_t, b_cb, biasL);
  attn_kernel<<<(int)N, 256, 0, stream>>>(nbr, mask, qB, kB, vB, biasL, oB);
  dim3 g512(512 / 64, (int)(N / 128));
  gemm_kernel<false, true, true, false><<<g512, 256, 0, stream>>>(
      oB, Wo_t, nullptr, features, out, nullptr, (int)N, 512, 256);
  ln512_kernel<<<(int)(N / 8), 256, 0, stream>>>(out, ln_up_s, ln_up_b,
                                                 yln, nullptr, (int)N);
  lp_mlp1_kernel<<<(int)(N * 1024 / 256), 256, 0, stream>>>(lp, W_lp1, b_lp1, lph);
  gemm_kernel<true, true, false, true><<<g512, 256, 0, stream>>>(
      lph, Wl2_t, b_lp2, yln, nullptr, yB, (int)N, 512, 1024);
  dim3 g2048(2048 / 64, (int)(N / 128));
  gemm_gu_kernel<<<g2048, 256, 0, stream>>>(yB, Wu_t, Wg_t, guB, (int)N, 2048, 512);
  gemm_kernel<true, true, true, false><<<g512, 256, 0, stream>>>(
      guB, Wo2_t, b_o2, out, out, nullptr, (int)N, 512, 2048);
}
